// self_attention_48593259987100
// MI455X (gfx1250) — matrix operand order repaired and checked by reading, not compiled, not run on hardware
//
#include <hip/hip_runtime.h>
#include <cstdint>

typedef __attribute__((ext_vector_type(16))) _Float16 v16h;
typedef __attribute__((ext_vector_type(8)))  float    v8f;

#define B_   8
#define D_   256
#define HW_  4096

// LDS row strides (in halves), padded to kill bank conflicts on fragment gathers
#define SK 264   // K tiles stored transposed: [kcol][d]
#define SQ 264   // Q tiles row-major:         [q][d]
#define SV 72    // V tile stored transposed:  [vd][k]
#define SP 72    // P tile row-major:          [q][k]

union FragU { v16h h; uint32_t u[8]; };

// A fragment (16x32 f16, M x K): lane L holds row M=L&15; VGPR v holds the
// packed K pair at  kbase + 2v + (v>=4 ? 8 : 0) + 8*(L>=16).
__device__ __forceinline__ v16h frag_a(const _Float16* rowptr, int kbase, int hi) {
  FragU f;
#pragma unroll
  for (int v = 0; v < 8; ++v) {
    int k = kbase + 2 * v + ((v & 4) ? 8 : 0) + 8 * hi;
    f.u[v] = *reinterpret_cast<const uint32_t*>(rowptr + k);
  }
  return f.h;
}

// B fragment (32x16 f16, K x N): lane L holds column N=L&15; VGPR v holds the
// packed K pair at  kbase + 2v + (v>=4 ? 8 : 0) + 8*(L>=16).  colptr points at the transposed
// LDS row for this column, so the pair is contiguous.
__device__ __forceinline__ v16h frag_b(const _Float16* colptr, int kbase, int hi) {
  FragU f;
#pragma unroll
  for (int v = 0; v < 8; ++v) {
    int k = kbase + 2 * v + ((v & 4) ? 8 : 0) + 8 * hi;
    f.u[v] = *reinterpret_cast<const uint32_t*>(colptr + k);
  }
  return f.h;
}

__device__ __forceinline__ v8f wmma16x16x32(v16h a, v16h b, v8f c) {
  return __builtin_amdgcn_wmma_f32_16x16x32_f16(false, a, false, b, (short)0, c,
                                                false, false);
}

// -------- Pass 1: per-(b,k) column max & sum-exp over the query axis --------
// grid (HW/64, B), block 128 (4 waves). Wave w owns k columns [k0+16w, +16).
__global__ __launch_bounds__(128)
void attn_colstats_kernel(const float* __restrict__ Kg,
                          const float* __restrict__ Qg,
                          float* __restrict__ Mws, float* __restrict__ Sws) {
  __shared__ _Float16 Kt[64 * SK];  // [kcol][d]
  __shared__ _Float16 Qs[16 * SQ];  // [q][d]

  const int tid = threadIdx.x;
  const int wave = tid >> 5, lane = tid & 31;
  const int ln = lane & 15, hi = lane >> 4;
  const int b = blockIdx.y;
  const int k0 = blockIdx.x * 64;
  const float* Kb = Kg + (size_t)b * D_ * HW_;
  const float* Qb = Qg + (size_t)b * D_ * HW_;

  // Stage this block's 64 K columns (all 256 d) once, transposed, as f16.
  for (int i = tid; i < 64 * D_; i += 128) {
    int kk = i & 63, d = i >> 6;
    Kt[kk * SK + d] = (_Float16)Kb[(size_t)d * HW_ + k0 + kk];
  }

  float m_run = -__builtin_inff();
  float s_run = 0.0f;

  const _Float16* arow = Qs + ln * SQ;
  const _Float16* bcol = Kt + (wave * 16 + ln) * SK;

  for (int qt = 0; qt < HW_ / 16; ++qt) {
    int q0 = qt * 16;
    for (int i = tid; i < 16 * D_; i += 128) {
      int qq = i & 15, d = i >> 4;
      Qs[qq * SQ + d] = (_Float16)Qb[(size_t)d * HW_ + q0 + qq];
    }
    __syncthreads();

    v8f acc = {};
#pragma unroll
    for (int kd = 0; kd < 8; ++kd)
      acc = wmma16x16x32(frag_a(arow, kd * 32, hi), frag_b(bcol, kd * 32, hi), acc);

    // Column-wise (over q) online max / sum-exp.  C layout: col N = lane&15,
    // rows M = r + 8*(lane>=16) across the 8 accumulator VGPRs.
    float tmax = acc[0];
#pragma unroll
    for (int r = 1; r < 8; ++r) tmax = fmaxf(tmax, acc[r]);
    tmax = fmaxf(tmax, __shfl_xor(tmax, 16, 32));
    float nm = fmaxf(m_run, tmax);
    float ls = 0.0f;
#pragma unroll
    for (int r = 0; r < 8; ++r) ls += __expf(acc[r] - nm);
    ls += __shfl_xor(ls, 16, 32);
    s_run = s_run * __expf(m_run - nm) + ls;
    m_run = nm;
    __syncthreads();  // Qs reused next iteration
  }

  if (hi == 0) {  // lane pair holds identical stats; low half writes
    int kcol = k0 + wave * 16 + ln;
    Mws[(size_t)b * HW_ + kcol] = m_run;
    Sws[(size_t)b * HW_ + kcol] = s_run;
  }
}

// -------- Pass 2: out[b,q,vd] = sum_k exp(s[q,k]-m_k)/sigma_k * V[k,vd] -----
// grid (HW/16, VD/64, B), block 128. Block: 16 q rows x 64 vd cols.
// Wave w owns k tile [16w,+16) of each 64-wide k step and vd cols [16w,+16).
__global__ __launch_bounds__(128)
void attn_out_kernel(const float* __restrict__ Kg,
                     const float* __restrict__ Qg,
                     const float* __restrict__ Vg,
                     const float* __restrict__ Mws,
                     const float* __restrict__ Sws,
                     float* __restrict__ outg) {
  __shared__ _Float16 Kt[64 * SK];  // [kcol][d]
  __shared__ _Float16 Qs[16 * SQ];  // [q][d]
  __shared__ _Float16 Vs[64 * SV];  // [vd][k]
  __shared__ _Float16 Ps[16 * SP];  // [q][k]

  const int tid = threadIdx.x;
  const int wave = tid >> 5, lane = tid & 31;
  const int ln = lane & 15, hi = lane >> 4;
  const int b = blockIdx.z;
  const int q0 = blockIdx.x * 16;
  const int vd0 = blockIdx.y * 64;
  const float* Kb = Kg + (size_t)b * D_ * HW_;
  const float* Qb = Qg + (size_t)b * D_ * HW_;
  const float* Vb = Vg + (size_t)b * D_ * HW_;

  for (int i = tid; i < 16 * D_; i += 128) {
    int qq = i & 15, d = i >> 4;
    Qs[qq * SQ + d] = (_Float16)Qb[(size_t)d * HW_ + q0 + qq];
  }

  v8f acc2 = {};
  const _Float16* arow = Qs + ln * SQ;
  const _Float16* bcol = Kt + (wave * 16 + ln) * SK;
  const _Float16* prow = Ps + ln * SP;
  const _Float16* vcol = Vs + (wave * 16 + ln) * SV;

  for (int ks = 0; ks < HW_ / 64; ++ks) {
    int k0 = ks * 64;
    for (int i = tid; i < 64 * D_; i += 128) {
      int kk = i & 63, d = i >> 6;
      Kt[kk * SK + d] = (_Float16)Kb[(size_t)d * HW_ + k0 + kk];
    }
    for (int i = tid; i < 64 * 64; i += 128) {
      int kk = i & 63, vl = i >> 6;
      Vs[vl * SV + kk] = (_Float16)Vb[(size_t)(vd0 + vl) * HW_ + k0 + kk];
    }
    if (ks + 1 < HW_ / 64)  // pull next K tile toward L2/L0
      __builtin_prefetch(Kb + (size_t)(tid >> 6) * HW_ + k0 + 64 + (tid & 63), 0, 1);
    __syncthreads();

    // scores tile [16 q x 16 k] for this wave's k slice (full d=256 contraction)
    v8f s = {};
#pragma unroll
    for (int kd = 0; kd < 8; ++kd)
      s = wmma16x16x32(frag_a(arow, kd * 32, hi), frag_b(bcol, kd * 32, hi), s);

    int kcol = k0 + wave * 16 + ln;
    float mk = Mws[(size_t)b * HW_ + kcol];
    float rs = 1.0f / Sws[(size_t)b * HW_ + kcol];
#pragma unroll
    for (int r = 0; r < 8; ++r) {
      float p = __expf(s[r] - mk) * rs;                // att[q, k]
      Ps[(r + 8 * hi) * SP + wave * 16 + ln] = (_Float16)p;
    }
    __syncthreads();  // all waves' P columns visible

    // out += P[16 x 64] * V[64 x 16]
#pragma unroll
    for (int kc = 0; kc < 64; kc += 32)
      acc2 = wmma16x16x32(frag_a(prow, kc, hi), frag_b(vcol, kc, hi), acc2);
    __syncthreads();  // Kt/Vs/Ps reused next iteration
  }

  // out[b, vd, q]: lane owns col N = vd0+16w+ln, rows q0 + r + 8*hi (contiguous)
  int col = vd0 + wave * 16 + ln;
  float* po = outg + (size_t)b * D_ * HW_ + (size_t)col * HW_ + q0 + hi * 8;
  float4 o0 = {acc2[0], acc2[1], acc2[2], acc2[3]};
  float4 o1 = {acc2[4], acc2[5], acc2[6], acc2[7]};
  *reinterpret_cast<float4*>(po) = o0;
  *reinterpret_cast<float4*>(po + 4) = o1;
}

extern "C" void kernel_launch(void* const* d_in, const int* in_sizes, int n_in,
                              void* d_out, int out_size, void* d_ws, size_t ws_size,
                              hipStream_t stream) {
  (void)in_sizes; (void)n_in; (void)out_size; (void)ws_size;
  const float* K = (const float*)d_in[0];
  const float* Q = (const float*)d_in[1];
  const float* V = (const float*)d_in[2];
  float* out = (float*)d_out;
  float* Mws = (float*)d_ws;               // [B, HW] column max
  float* Sws = Mws + (size_t)B_ * HW_;     // [B, HW] column sum-exp

  dim3 g1(HW_ / 64, B_);
  attn_colstats_kernel<<<g1, 128, 0, stream>>>(K, Q, Mws, Sws);

  dim3 g2(HW_ / 16, D_ / 64, B_);
  attn_out_kernel<<<g2, 128, 0, stream>>>(K, Q, V, Mws, Sws, out);
}
